// Attn_67894843015671
// MI455X (gfx1250) — matrix operand order repaired and checked by reading, not compiled, not run on hardware
//
#include <hip/hip_runtime.h>
#include <hip/hip_bf16.h>

// ---------------------------------------------------------------------------
// Problem constants (from reference):
//   B=4, C=128, T=512, F=128, H=4, D=32, CH=32, DF = D*F = 4096, BH = B*H = 16
// ---------------------------------------------------------------------------
#define B_ 4
#define C_ 128
#define T_ 512
#define F_ 128
#define H_ 4
#define D_ 32
#define CH_ 32
#define DF_ 4096
#define BH_ 16

typedef __attribute__((ext_vector_type(16))) _Float16 v16h;
typedef __attribute__((ext_vector_type(8)))  _Float16 v8h;
typedef __attribute__((ext_vector_type(8)))  float    v8f;

// Wave-uniform broadcast: lets the compiler keep tile coordinates in SGPRs,
// turning tile-selection branches into scalar branches (no exec juggling).
__device__ __forceinline__ int uni(int v) {
  return __builtin_amdgcn_readfirstlane(v);
}

// ---------------------------------------------------------------------------
// WMMA fragment loaders (layouts per CDNA5 ISA 7.12.2, wave32)
//
// A (16x32 f16, MxK): lane holds row M = lane&15; hi = lane>>4.
//   a[j]   = K(k0 + hi*8 + j),       j = 0..7   (16B contiguous)
//   a[8+j] = K(k0 + 16 + hi*8 + j),  j = 0..7   (16B contiguous)
//
// B (32x16 f16, KxN) loaded from B-transposed storage [N][K] (K contiguous):
//   lane holds col N = lane&15; b[j] = K(k0 + hi*8 + j), b[8+j] = K(k0 + 16 + hi*8 + j), j = 0..7
//
// C/D (16x16 f32): c[r] -> (M = r + 8*hi, N = lane&15)
// ---------------------------------------------------------------------------
__device__ __forceinline__ v16h load_a_frag(const _Float16* base, int rowStride,
                                            int m0, int k0, int lane) {
  const int row = lane & 15;
  const int hi  = (lane >> 4) & 1;
  const _Float16* p = base + (size_t)(m0 + row) * (size_t)rowStride + k0;
  union { v16h v; v8h h[2]; } u;
  u.h[0] = *(const v8h*)(p + hi * 8);
  u.h[1] = *(const v8h*)(p + 16 + hi * 8);
  return u.v;
}

__device__ __forceinline__ v16h load_bt_frag(const _Float16* base, int rowStride,
                                             int n0, int k0, int lane) {
  const int col = lane & 15;
  const int hi  = (lane >> 4) & 1;
  const _Float16* p = base + (size_t)(n0 + col) * (size_t)rowStride + k0;
  union { v16h v; v8h h[2]; } u;
  u.h[0] = *(const v8h*)(p + hi * 8);
  u.h[1] = *(const v8h*)(p + 16 + hi * 8);
  return u.v;
}

__device__ __forceinline__ v8f wmma16(v16h a, v16h b, v8f c) {
  return __builtin_amdgcn_wmma_f32_16x16x32_f16(
      /*neg_a=*/false, a, /*neg_b=*/false, b,
      /*c_mod=*/(short)0, c, /*reuse_a=*/false, /*reuse_b=*/false);
}

__device__ __forceinline__ v8f zero8() {
  v8f z;
#pragma unroll
  for (int i = 0; i < 8; ++i) z[i] = 0.0f;
  return z;
}

// ---------------------------------------------------------------------------
// K0: convert weights to f16.
//   W16[384][128]: rows 0-127 = qw (h*32+d, c), 128-255 = kw, 256-383 = vw
//   FW16[128][128] = fw (o, c)
// ---------------------------------------------------------------------------
__global__ void prep_weights_kernel(const float* __restrict__ qw,
                                    const float* __restrict__ kw,
                                    const float* __restrict__ vw,
                                    const float* __restrict__ fw,
                                    _Float16* __restrict__ W16,
                                    _Float16* __restrict__ FW16) {
  int idx = blockIdx.x * blockDim.x + threadIdx.x;  // 0 .. 65535
  const int NW = 384 * 128;                          // 49152
  if (idx < NW) {
    float v;
    if (idx < 16384)       v = qw[idx];
    else if (idx < 32768)  v = kw[idx - 16384];
    else                   v = vw[idx - 32768];
    W16[idx] = (_Float16)v;
  } else {
    int j = idx - NW;
    if (j < 128 * 128) FW16[j] = (_Float16)fw[j];
  }
}

// ---------------------------------------------------------------------------
// K1: transpose x[b,c,t,f] (f32) -> XT16[b,t,f,c] (f16), LDS-tiled 32x32.
// grid = (16, B*T), block = (32, 8)
// ---------------------------------------------------------------------------
__global__ void transpose_x_kernel(const float* __restrict__ x,
                                   _Float16* __restrict__ XT) {
  __shared__ float tile[32][33];
  const int bt = blockIdx.y;              // b*T + t
  const int b  = bt >> 9;
  const int t  = bt & 511;
  const int c0 = (blockIdx.x >> 2) * 32;  // 4 c-tiles
  const int f0 = (blockIdx.x & 3) * 32;   // 4 f-tiles
#pragma unroll
  for (int k = 0; k < 4; ++k) {
    int c = c0 + threadIdx.y + k * 8;
    int f = f0 + threadIdx.x;
    tile[threadIdx.y + k * 8][threadIdx.x] =
        x[(((size_t)(b * C_ + c) * T_) + t) * F_ + f];
  }
  __syncthreads();
#pragma unroll
  for (int k = 0; k < 4; ++k) {
    int f = f0 + threadIdx.y + k * 8;
    int c = c0 + threadIdx.x;
    XT[((size_t)bt * F_ + f) * C_ + c] =
        (_Float16)tile[threadIdx.x][threadIdx.y + k * 8];
  }
}

// ---------------------------------------------------------------------------
// K2: QKV projection. Per (b,t): Out[384,128] = W16[384,128] @ xslice[128,128],
// +bias, ReLU. Q/K -> [b,h,t, d*F+f] f16; V -> VT[b,h, e*F+f, t] f16.
// grid = B*T (2048), block = 256 (8 waves). Wave w owns m-tiles [3w, 3w+2].
// wid is readfirstlane'd -> region branch is a scalar branch.
// ---------------------------------------------------------------------------
__global__ void qkv_kernel(const float* __restrict__ qb,
                           const float* __restrict__ kb,
                           const float* __restrict__ vb,
                           const _Float16* __restrict__ W16,
                           const _Float16* __restrict__ XT,
                           _Float16* __restrict__ Qf,
                           _Float16* __restrict__ Kf,
                           _Float16* __restrict__ VT) {
  const int bt   = blockIdx.x;
  const int b    = bt >> 9;
  const int t    = bt & 511;
  const int wid  = uni(threadIdx.x >> 5);
  const int lane = threadIdx.x & 31;
  const int col16 = lane & 15;
  const int hi    = (lane >> 4) & 1;
  const _Float16* Bbase = XT + (size_t)bt * (F_ * C_);  // rows f, stride C

  for (int mi = wid * 3; mi < wid * 3 + 3; ++mi) {
    v16h aF[4];
#pragma unroll
    for (int kc = 0; kc < 4; ++kc)
      aF[kc] = load_a_frag(W16, C_, mi * 16, kc * 32, lane);

#pragma unroll 2
    for (int ni = 0; ni < 8; ++ni) {
      v16h bF[4];
#pragma unroll
      for (int kc = 0; kc < 4; ++kc)
        bF[kc] = load_bt_frag(Bbase, C_, ni * 16, kc * 32, lane);
      v8f acc = zero8();
#pragma unroll
      for (int kc = 0; kc < 4; ++kc)
        acc = wmma16(aF[kc], bF[kc], acc);

      const int n = ni * 16 + col16;  // f
      if (mi < 8) {                   // Q region (rows 0-127) -- scalar branch
        const int m0 = mi * 16 + 8 * hi;
#pragma unroll
        for (int r = 0; r < 8; ++r) {
          const int m = m0 + r;
          const float v = fmaxf(acc[r] + qb[m], 0.0f);
          const int h = m >> 5, d = m & 31;
          Qf[(((size_t)(b * H_ + h) * T_) + t) * DF_ + d * F_ + n] = (_Float16)v;
        }
      } else if (mi < 16) {           // K region (rows 128-255)
        const int m0 = mi * 16 - 128 + 8 * hi;
#pragma unroll
        for (int r = 0; r < 8; ++r) {
          const int m = m0 + r;
          const float v = fmaxf(acc[r] + kb[m], 0.0f);
          const int h = m >> 5, d = m & 31;
          Kf[(((size_t)(b * H_ + h) * T_) + t) * DF_ + d * F_ + n] = (_Float16)v;
        }
      } else {                        // V region (rows 256-383), transposed out
        const int m0 = mi * 16 - 256 + 8 * hi;
#pragma unroll
        for (int r = 0; r < 8; ++r) {
          const int m = m0 + r;
          const float v = fmaxf(acc[r] + vb[m], 0.0f);
          const int h = m >> 5, e = m & 31;
          VT[((size_t)(b * H_ + h) * DF_ + e * F_ + n) * T_ + t] = (_Float16)v;
        }
      }
    }
  }
}

// ---------------------------------------------------------------------------
// K3: logits[bh][t][s] = (1/64) * sum_feat Qf[t,feat] * Kf[s,feat]
// 512x512x4096 per bh. 64x64 tile per wave (4x4 accs, 16 WMMA per K-chunk).
// grid = (16, BH), block = 128 (4 waves)
// ---------------------------------------------------------------------------
__global__ void logits_kernel(const _Float16* __restrict__ Qf,
                              const _Float16* __restrict__ Kf,
                              float* __restrict__ logits) {
  const int bh   = blockIdx.y;
  const int wid  = uni(threadIdx.x >> 5);
  const int lane = threadIdx.x & 31;
  const int tileId = blockIdx.x * 4 + wid;  // 0..63 (SGPR)
  const int m64 = tileId >> 3;
  const int n64 = tileId & 7;
  const _Float16* Qb = Qf + (size_t)bh * T_ * DF_;
  const _Float16* Kb = Kf + (size_t)bh * T_ * DF_;

  v8f acc[4][4];
#pragma unroll
  for (int i = 0; i < 4; ++i)
#pragma unroll
    for (int j = 0; j < 4; ++j) acc[i][j] = zero8();

  for (int kc = 0; kc < DF_ / 32; ++kc) {  // 128 K-chunks
    const int k0 = kc * 32;
    v16h aF[4], bF[4];
#pragma unroll
    for (int i = 0; i < 4; ++i)
      aF[i] = load_a_frag(Qb, DF_, m64 * 64 + i * 16, k0, lane);
#pragma unroll
    for (int j = 0; j < 4; ++j)
      bF[j] = load_bt_frag(Kb, DF_, n64 * 64 + j * 16, k0, lane);
#pragma unroll
    for (int i = 0; i < 4; ++i)
#pragma unroll
      for (int j = 0; j < 4; ++j)
        acc[i][j] = wmma16(aF[i], bF[j], acc[i][j]);
  }

  const float scale = 0.015625f;  // 1/sqrt(4096)
  const int col16 = lane & 15;
  const int hi    = (lane >> 4) & 1;
#pragma unroll
  for (int i = 0; i < 4; ++i)
#pragma unroll
    for (int j = 0; j < 4; ++j)
#pragma unroll
      for (int r = 0; r < 8; ++r) {
        const int t = m64 * 64 + i * 16 + r + 8 * hi;
        const int s = n64 * 64 + j * 16 + col16;
        logits[((size_t)bh * T_ + t) * T_ + s] = acc[i][j][r] * scale;
      }
}

// ---------------------------------------------------------------------------
// K4: softmax over s. One wave32 per row of 512. attn written as f16.
// grid = BH*T/8 = 1024, block = 256
// ---------------------------------------------------------------------------
__global__ void softmax_kernel(const float* __restrict__ logits,
                               _Float16* __restrict__ attn) {
  const int wid  = uni(threadIdx.x >> 5);
  const int lane = threadIdx.x & 31;
  const int row  = blockIdx.x * 8 + wid;  // 0 .. BH*T-1 (SGPR)
  const float* lp = logits + (size_t)row * T_;

  float vals[16];
  float m = -__builtin_huge_valf();
#pragma unroll
  for (int j = 0; j < 16; ++j) {
    vals[j] = lp[lane + j * 32];
    m = fmaxf(m, vals[j]);
  }
#pragma unroll
  for (int off = 16; off >= 1; off >>= 1)
    m = fmaxf(m, __shfl_xor(m, off, 32));

  float s = 0.0f;
#pragma unroll
  for (int j = 0; j < 16; ++j) {
    vals[j] = expf(vals[j] - m);
    s += vals[j];
  }
#pragma unroll
  for (int off = 16; off >= 1; off >>= 1)
    s += __shfl_xor(s, off, 32);

  const float inv = 1.0f / s;
  _Float16* ap = attn + (size_t)row * T_;
#pragma unroll
  for (int j = 0; j < 16; ++j)
    ap[lane + j * 32] = (_Float16)(vals[j] * inv);
}

// ---------------------------------------------------------------------------
// K5: O[bh][t][n] = sum_s attn[t,s] * V[s,n] ; 512x4096x512 per bh.
// 64x64 tile per wave. Output scattered into OT16[b, t, f, c] with c=h*32+e
// (c is constant within a wave tile; OT (64MB) lives in the 192MB L2, so the
// 2-byte scatter is absorbed by L2 write combining).
// grid = (128, BH), block = 128 (4 waves)
// ---------------------------------------------------------------------------
__global__ void ogemm_kernel(const _Float16* __restrict__ attn,
                             const _Float16* __restrict__ VT,
                             _Float16* __restrict__ OT) {
  const int bh   = blockIdx.y;
  const int b    = bh >> 2;
  const int h    = bh & 3;
  const int wid  = uni(threadIdx.x >> 5);
  const int lane = threadIdx.x & 31;
  const int tileId = blockIdx.x * 4 + wid;  // 0..511 (SGPR)
  const int m64 = tileId >> 6;              // 0..7   (t blocks)
  const int n64 = tileId & 63;              // 0..63  (feature blocks)
  const _Float16* Ab = attn + (size_t)bh * T_ * T_;  // rows t, stride T
  const _Float16* Bb = VT + (size_t)bh * DF_ * T_;   // rows n, stride T

  v8f acc[4][4];
#pragma unroll
  for (int i = 0; i < 4; ++i)
#pragma unroll
    for (int j = 0; j < 4; ++j) acc[i][j] = zero8();

  for (int kc = 0; kc < T_ / 32; ++kc) {  // 16 K-chunks over s
    const int k0 = kc * 32;
    v16h aF[4], bF[4];
#pragma unroll
    for (int i = 0; i < 4; ++i)
      aF[i] = load_a_frag(Ab, T_, m64 * 64 + i * 16, k0, lane);
#pragma unroll
    for (int j = 0; j < 4; ++j)
      bF[j] = load_bt_frag(Bb, T_, n64 * 64 + j * 16, k0, lane);
#pragma unroll
    for (int i = 0; i < 4; ++i)
#pragma unroll
      for (int j = 0; j < 4; ++j)
        acc[i][j] = wmma16(aF[i], bF[j], acc[i][j]);
  }

  const int col16 = lane & 15;
  const int hi    = (lane >> 4) & 1;
#pragma unroll
  for (int i = 0; i < 4; ++i)
#pragma unroll
    for (int j = 0; j < 4; ++j)
#pragma unroll
      for (int r = 0; r < 8; ++r) {
        const int t = m64 * 64 + i * 16 + r + 8 * hi;
        const int n = n64 * 64 + j * 16 + col16;  // e*F + f
        const int e = n >> 7, f = n & 127;
        const int c = h * CH_ + e;
        OT[(((size_t)(b * T_ + t)) * F_ + f) * C_ + c] = (_Float16)acc[i][j][r];
      }
}

// ---------------------------------------------------------------------------
// K6: out[b,o,t,f] = relu(sum_c fw[o,c]*O[b,c,t,f] + fb[o]) + x[b,o,t,f]
// Per (b,t): FW16[128,128] @ OTslice^T. grid = B*T, block = 256 (8 waves).
// ---------------------------------------------------------------------------
__global__ void final_kernel(const float* __restrict__ x,
                             const float* __restrict__ fb,
                             const _Float16* __restrict__ FW16,
                             const _Float16* __restrict__ OT,
                             float* __restrict__ out) {
  const int bt   = blockIdx.x;
  const int b    = bt >> 9;
  const int t    = bt & 511;
  const int wid  = uni(threadIdx.x >> 5);
  const int lane = threadIdx.x & 31;
  const int col16 = lane & 15;
  const int hi    = (lane >> 4) & 1;
  const _Float16* Bbase = OT + (size_t)bt * (F_ * C_);  // rows f, stride C
  const int mi = wid;  // one m-tile strip per wave (SGPR)

  v16h aF[4];
#pragma unroll
  for (int kc = 0; kc < 4; ++kc)
    aF[kc] = load_a_frag(FW16, C_, mi * 16, kc * 32, lane);

#pragma unroll 2
  for (int ni = 0; ni < 8; ++ni) {
    v16h bF[4];
#pragma unroll
    for (int kc = 0; kc < 4; ++kc)
      bF[kc] = load_bt_frag(Bbase, C_, ni * 16, kc * 32, lane);
    v8f acc = zero8();
#pragma unroll
    for (int kc = 0; kc < 4; ++kc)
      acc = wmma16(aF[kc], bF[kc], acc);

    const int f = ni * 16 + col16;
#pragma unroll
    for (int r = 0; r < 8; ++r) {
      const int o = mi * 16 + r + 8 * hi;
      const size_t idx = (((size_t)(b * C_ + o) * T_) + t) * F_ + f;
      out[idx] = fmaxf(acc[r] + fb[o], 0.0f) + x[idx];
    }
  }
}

// ---------------------------------------------------------------------------
// Host launcher
// ---------------------------------------------------------------------------
extern "C" void kernel_launch(void* const* d_in, const int* in_sizes, int n_in,
                              void* d_out, int out_size, void* d_ws, size_t ws_size,
                              hipStream_t stream) {
  const float* x  = (const float*)d_in[0];
  const float* qw = (const float*)d_in[1];
  const float* qb = (const float*)d_in[2];
  const float* kw = (const float*)d_in[3];
  const float* kb = (const float*)d_in[4];
  const float* vw = (const float*)d_in[5];
  const float* vb = (const float*)d_in[6];
  const float* fw = (const float*)d_in[7];
  const float* fb = (const float*)d_in[8];
  float* out = (float*)d_out;

  char* ws = (char*)d_ws;
  // Workspace layout (256B-aligned offsets)
  const size_t o_W16  = 0;                            //   98,304 B
  const size_t o_FW16 = o_W16  + 384 * 128 * 2;       //   32,768 B
  const size_t o_XT   = o_FW16 + 128 * 128 * 2;       // 64 MiB (reused as OT)
  const size_t sz_XT  = (size_t)B_ * T_ * F_ * C_ * 2;
  const size_t o_Qf   = o_XT + sz_XT;                 // 64 MiB
  const size_t sz_QKV = (size_t)BH_ * T_ * DF_ * 2;
  const size_t o_Kf   = o_Qf + sz_QKV;                // 64 MiB
  const size_t o_VT   = o_Kf + sz_QKV;                // 64 MiB
  const size_t o_log  = o_VT + sz_QKV;                // 16 MiB
  const size_t o_attn = o_log + (size_t)BH_ * T_ * T_ * 4;  // 8 MiB

  _Float16* W16  = (_Float16*)(ws + o_W16);
  _Float16* FW16 = (_Float16*)(ws + o_FW16);
  _Float16* XT   = (_Float16*)(ws + o_XT);
  _Float16* OT   = (_Float16*)(ws + o_XT);  // alias: XT dead after qkv_kernel
  _Float16* Qf   = (_Float16*)(ws + o_Qf);
  _Float16* Kf   = (_Float16*)(ws + o_Kf);
  _Float16* VT   = (_Float16*)(ws + o_VT);
  float*    logits = (float*)(ws + o_log);
  _Float16* attn   = (_Float16*)(ws + o_attn);

  // K0: weights -> f16 (65536 elements)
  prep_weights_kernel<<<256, 256, 0, stream>>>(qw, kw, vw, fw, W16, FW16);
  // K1: x -> XT16 (transposed, f16)
  transpose_x_kernel<<<dim3(16, B_ * T_), dim3(32, 8), 0, stream>>>(x, XT);
  // K2: QKV projection (WMMA)
  qkv_kernel<<<B_ * T_, 256, 0, stream>>>(qb, kb, vb, W16, XT, Qf, Kf, VT);
  // K3: logits (WMMA)
  logits_kernel<<<dim3(16, BH_), 128, 0, stream>>>(Qf, Kf, logits);
  // K4: softmax
  softmax_kernel<<<BH_ * T_ / 8, 256, 0, stream>>>(logits, attn);
  // K5: attn @ V (WMMA), scatter to OT
  ogemm_kernel<<<dim3(128, BH_), 128, 0, stream>>>(attn, VT, OT);
  // K6: final projection + ReLU + residual (WMMA)
  final_kernel<<<B_ * T_, 256, 0, stream>>>(x, fb, FW16, OT, out);

  (void)in_sizes; (void)n_in; (void)out_size; (void)ws_size;
}
